// GPTJSelfAttention_77257871720646
// MI455X (gfx1250) — matrix operand order repaired and checked by reading, not compiled, not run on hardware
//
#include <hip/hip_runtime.h>
#include <hip/hip_bf16.h>

// ---------------------------------------------------------------------------
// GPT-J self-attention for MI455X (gfx1250): bf16 WMMA everywhere.
//   q/k/v = X @ W.T  (NT GEMM, v_wmma_f32_16x16x32_bf16)
//   rope + transpose, flash-attention with TDM double-buffered K/V tiles,
//   out = ctx @ Wo.T
// ---------------------------------------------------------------------------

typedef __attribute__((ext_vector_type(16))) __bf16 v16bf;
typedef __attribute__((ext_vector_type(8)))  float  v8f;
typedef __attribute__((ext_vector_type(4))) unsigned int v4u;
typedef __attribute__((ext_vector_type(8))) int          v8i;
typedef __attribute__((ext_vector_type(4))) int          v4i;

#define NHEADS  16
#define HDIM    256
#define DMODEL  4096
#define SEQ     2048
#define BATCH   2

__device__ __forceinline__ unsigned short f2bf(float f) {
  union { float f; unsigned int u; } x; x.f = f;
  unsigned int u = x.u;
  return (unsigned short)((u + 0x7FFFu + ((u >> 16) & 1u)) >> 16);
}
__device__ __forceinline__ float bf2f(unsigned short s) {
  union { unsigned int u; float f; } x; x.u = ((unsigned int)s) << 16;
  return x.f;
}

union FragB {
  v16bf v;
  unsigned short u[16];
  uint4 q[2];
};

__device__ __forceinline__ v8f wmma_bf16(const FragB& a, const FragB& b, v8f c) {
  return __builtin_amdgcn_wmma_f32_16x16x32_bf16(false, a.v, false, b.v,
                                                 (short)0, c, false, false);
}

// ---------------------------------------------------------------------------
// TDM: async DMA of a 2D bf16 tile (tile_d0 x tile_d1, row stride stride0
// elements) from global memory into LDS at byte offset lds_off.
// Packs D# group0 (count/lds_addr/global_addr/type=2) and group1
// (data_size=2B, tensor dims, tile dims, dim0 stride) per CDNA5 ISA 8.3/8.4.
// This toolchain exposes the 6-arg builtin: (g0, g1, g2, g3, g4, cpol).
// ---------------------------------------------------------------------------
__device__ __forceinline__ void tdm_load_2d(unsigned lds_off, const void* gaddr,
                                            unsigned tile_d0, unsigned tile_d1,
                                            unsigned tensor_d0, unsigned tensor_d1,
                                            unsigned long long stride0) {
  unsigned long long ga = (unsigned long long)gaddr;
  v4u g0;
  g0[0] = 1u;                                        // count=1 (valid descriptor)
  g0[1] = lds_off;                                   // lds_addr (bytes)
  g0[2] = (unsigned)(ga & 0xffffffffu);              // global_addr[31:0]
  g0[3] = (unsigned)((ga >> 32) & 0x01ffffffu)       // global_addr[56:32]
        | (2u << 30);                                // type=2 ("image")
  v8i g1;
  g1[0] = (int)(1u << 16);                           // data_size=1 -> 2 bytes
  g1[1] = (int)((tensor_d0 & 0xffffu) << 16);        // tensor_dim0[15:0]
  g1[2] = (int)((tensor_d0 >> 16) | ((tensor_d1 & 0xffffu) << 16));
  g1[3] = (int)((tensor_d1 >> 16) | ((tile_d0 & 0xffffu) << 16));
  g1[4] = (int)(tile_d1 & 0xffffu);                  // tile_dim1, tile_dim2=0
  g1[5] = (int)(unsigned)(stride0 & 0xffffffffu);    // dim0_stride[31:0]
  g1[6] = (int)(unsigned)((stride0 >> 32) & 0xffffu);// dim0_stride[47:32]
  g1[7] = 0;
  v4i z4 = (v4i)0;
  v8i z8 = (v8i)0;
  __builtin_amdgcn_tensor_load_to_lds(g0, g1, z4, z4, z8, 0);
}

// ---------------------------------------------------------------------------
// fp32 -> bf16 elementwise cast
// ---------------------------------------------------------------------------
__global__ void cast_bf16_kernel(const float* __restrict__ in,
                                 unsigned short* __restrict__ out, long n) {
  long i = (long)blockIdx.x * blockDim.x + threadIdx.x;
  long stride = (long)gridDim.x * blockDim.x;
  for (; i < n; i += stride) out[i] = f2bf(in[i]);
}

// ---------------------------------------------------------------------------
// C[M,N] = A[M,K] * B[N,K]^T   (A,B bf16 row-major; C fp32 or bf16)
// block = 256 threads = 8 waves (4 wave-rows x 2 wave-cols)
// block tile 128x128; each wave: 32x64 = 2x4 tiles of 16x16
// ---------------------------------------------------------------------------
__global__ __launch_bounds__(256)
void gemm_nt_bf16(const unsigned short* __restrict__ A,
                  const unsigned short* __restrict__ B,
                  float* __restrict__ Cf,
                  unsigned short* __restrict__ Cb,
                  int M, int N, int K) {
  const int lane = threadIdx.x & 31;
  const int wave = threadIdx.x >> 5;
  const int r = lane & 15;
  const int h = lane >> 4;
  const int wr = wave >> 1;        // 0..3
  const int wc = wave & 1;         // 0..1
  const int m0 = blockIdx.y * 128 + wr * 32;
  const int n0 = blockIdx.x * 128 + wc * 64;

  v8f acc[2][4];
  for (int i = 0; i < 2; ++i)
    for (int j = 0; j < 4; ++j) acc[i][j] = (v8f){};

  for (int k0 = 0; k0 < K; k0 += 32) {
    FragB a[2], b[4];
    // A fragment: lane (r,h) holds row m0+i*16+r, K chunks h*8.. and 16+h*8..
#pragma unroll
    for (int i = 0; i < 2; ++i) {
      const unsigned short* p = A + (size_t)(m0 + i * 16 + r) * K + k0 + h * 8;
      a[i].q[0] = *(const uint4*)p;
      a[i].q[1] = *(const uint4*)(p + 16);
    }
    // B fragment: lane (r,h) holds column n0+j*16+r, K chunks h*8.. and 16+h*8..
#pragma unroll
    for (int j = 0; j < 4; ++j) {
      const unsigned short* p = B + (size_t)(n0 + j * 16 + r) * K + k0 + h * 8;
      b[j].q[0] = *(const uint4*)p;
      b[j].q[1] = *(const uint4*)(p + 16);
    }
    if (k0 + 32 < K) {   // pull next K tile toward the WGP (global_prefetch_b8)
      __builtin_prefetch(A + (size_t)(m0 + r) * K + k0 + 32, 0, 1);
      __builtin_prefetch(B + (size_t)(n0 + r) * K + k0 + 32, 0, 1);
    }
#pragma unroll
    for (int i = 0; i < 2; ++i)
#pragma unroll
      for (int j = 0; j < 4; ++j)
        acc[i][j] = wmma_bf16(a[i], b[j], acc[i][j]);
  }

  // C layout: VGPR e -> row m0+i*16+e+8*h, col n0+j*16+r
#pragma unroll
  for (int i = 0; i < 2; ++i)
#pragma unroll
    for (int j = 0; j < 4; ++j)
#pragma unroll
      for (int e = 0; e < 8; ++e) {
        int row = m0 + i * 16 + e + 8 * h;
        int col = n0 + j * 16 + r;
        float val = acc[i][j][e];
        if (Cb) Cb[(size_t)row * N + col] = f2bf(val);
        else    Cf[(size_t)row * N + col] = val;
      }
}

// ---------------------------------------------------------------------------
// RoPE (first 64 dims) + [B,S,H*hd] -> [B,H,S,hd] transpose, bf16 -> bf16
// ---------------------------------------------------------------------------
__global__ void rope_transpose_kernel(const unsigned short* __restrict__ src,
                                      unsigned short* __restrict__ dst,
                                      long total) {
  long i = (long)blockIdx.x * blockDim.x + threadIdx.x;
  if (i >= total) return;
  int d  = (int)(i & 255);
  long t = i >> 8;
  int hh = (int)(t & 15);
  t >>= 4;                 // t = b*SEQ + s
  int s = (int)(t % SEQ);
  int b = (int)(t / SEQ);

  float x = bf2f(src[i]);  // flat src index == i by construction
  float outv;
  if (d < 64) {
    int j = d >> 1;
    float ang = (float)s * powf(10000.0f, -(float)(2 * j) / 64.0f);
    float c = cosf(ang), sn = sinf(ang);
    float xp = bf2f(src[i ^ 1]);            // partner within the pair
    outv = (d & 1) ? (x * c + xp * sn) : (x * c - xp * sn);
  } else {
    outv = x;
  }
  size_t didx = (((size_t)(b * NHEADS + hh) * SEQ + s)) * HDIM + d;
  dst[didx] = f2bf(outv);
}

// [B,S,H*hd] -> [B,H,hd,S]  (V transposed so P*V B-fragments are contiguous)
__global__ void vtrans_kernel(const unsigned short* __restrict__ src,
                              unsigned short* __restrict__ dst, long total) {
  long i = (long)blockIdx.x * blockDim.x + threadIdx.x;
  if (i >= total) return;
  int d  = (int)(i & 255);
  long t = i >> 8;
  int hh = (int)(t & 15);
  t >>= 4;
  int s = (int)(t % SEQ);
  int b = (int)(t / SEQ);
  size_t didx = ((size_t)(b * NHEADS + hh) * HDIM + d) * SEQ + s;
  dst[didx] = src[i];
}

// ---------------------------------------------------------------------------
// Flash attention, one wave32 per workgroup, 16 query rows per wave.
// K/V tiles are DMA'd into LDS by the Tensor Data Mover, double-buffered:
// the TDM for chunk i+1 runs behind the 32 WMMAs of chunk i
// (s_wait_tensorcnt 2 = wait only for the two oldest descriptors).
// Qt/Kt: [B,H,S,hd] bf16; Vt: [B,H,hd,S] bf16; ctx: [B,S,D] bf16.
// ---------------------------------------------------------------------------
__global__ __launch_bounds__(32)
void flash_attn_kernel(const unsigned short* __restrict__ Qt,
                       const unsigned short* __restrict__ Kt,
                       const unsigned short* __restrict__ Vt,
                       unsigned short* __restrict__ ctx,
                       int S) {
  __shared__ __align__(16) unsigned short k_tile[2][32 * HDIM];  // [key][d]
  __shared__ __align__(16) unsigned short v_tile[2][HDIM * 32];  // [d][key]
  __shared__ __align__(16) unsigned short p_lds[16 * 32];

  const int lane = threadIdx.x & 31;
  const int r = lane & 15;
  const int h = lane >> 4;
  const int q0 = blockIdx.x * 16;
  const int hh = blockIdx.y;
  const int b  = blockIdx.z;
  const int bh = b * NHEADS + hh;

  const unsigned short* Q  = Qt + (size_t)bh * S * HDIM;
  const unsigned short* Kp = Kt + (size_t)bh * S * HDIM;
  const unsigned short* Vp = Vt + (size_t)bh * HDIM * S;

  // Resident Q fragments: 8 chunks of K=32 covering hd=256
  FragB qf[8];
#pragma unroll
  for (int c = 0; c < 8; ++c) {
    const unsigned short* p = Q + (size_t)(q0 + r) * HDIM + c * 32 + h * 8;
    qf[c].q[0] = *(const uint4*)p;
    qf[c].q[1] = *(const uint4*)(p + 16);
  }

  v8f o[16];
#pragma unroll
  for (int nt = 0; nt < 16; ++nt) o[nt] = (v8f){};
  float m_i[8], l_i[8];
#pragma unroll
  for (int i = 0; i < 8; ++i) { m_i[i] = -1.0e30f; l_i[i] = 0.0f; }

  const float scale = 0.0625f;             // 1/sqrt(256)
  const float NEG   = -6.25e7f;            // -1e9 * scale
  const int   kend  = q0 + 16;             // causal key bound (exclusive)

  // Prologue: TDM chunk 0 into buffer 0.
  tdm_load_2d((unsigned)(unsigned long long)&k_tile[0][0], Kp,
              HDIM, 32, HDIM, (unsigned)S, (unsigned long long)HDIM);
  tdm_load_2d((unsigned)(unsigned long long)&v_tile[0][0], Vp,
              32, HDIM, (unsigned)S, HDIM, (unsigned long long)S);

  for (int k0 = 0, it = 0; k0 < kend; k0 += 32, ++it) {
    const int  buf      = it & 1;
    const bool havenext = (k0 + 32) < kend;

    // WAR safety: all LDS reads from the buffer we are about to refill are done.
    asm volatile("s_wait_dscnt 0x0" ::: "memory");
    if (havenext) {
      tdm_load_2d((unsigned)(unsigned long long)&k_tile[buf ^ 1][0],
                  Kp + (size_t)(k0 + 32) * HDIM,
                  HDIM, 32, HDIM, (unsigned)S, (unsigned long long)HDIM);
      tdm_load_2d((unsigned)(unsigned long long)&v_tile[buf ^ 1][0],
                  Vp + (k0 + 32),
                  32, HDIM, (unsigned)S, HDIM, (unsigned long long)S);
      __builtin_amdgcn_s_wait_tensorcnt(2);   // chunk i's 2 descriptors done
    } else {
      __builtin_amdgcn_s_wait_tensorcnt(0);
    }
    const unsigned short* kt_p = &k_tile[buf][0];
    const unsigned short* vt_p = &v_tile[buf][0];
    // Launder pointers: TDM wrote this LDS behind the compiler's back.
    asm volatile("" : "+v"(kt_p), "+v"(vt_p) :: "memory");

    // ---- S = Q K^T over 32 keys: two 16x16 score tiles -------------------
    // second tile only if any of keys k0+16..k0+31 can be <= some query row
    const bool tile1 = (k0 + 16) < kend;
    v8f s0 = (v8f){}, s1 = (v8f){};
#pragma unroll
    for (int c = 0; c < 8; ++c) {
      FragB kbf;
      const unsigned short* pa = kt_p + (size_t)r * HDIM + c * 32 + h * 8;
      kbf.q[0] = *(const uint4*)pa;
      kbf.q[1] = *(const uint4*)(pa + 16);
      s0 = wmma_bf16(qf[c], kbf, s0);
      if (tile1) {
        const unsigned short* pb = kt_p + (size_t)(16 + r) * HDIM + c * 32 + h * 8;
        kbf.q[0] = *(const uint4*)pb;
        kbf.q[1] = *(const uint4*)(pb + 16);
        s1 = wmma_bf16(qf[c], kbf, s1);
      }
    }

    // ---- online softmax (rows live on 16-lane halves) --------------------
    float alpha8[8];
#pragma unroll
    for (int i = 0; i < 8; ++i) {
      int row = q0 + i + 8 * h;
      int c0 = k0 + r, c1 = k0 + 16 + r;
      float v0 = (c0 <= row) ? s0[i] * scale : NEG;
      float v1 = (tile1 && c1 <= row) ? s1[i] * scale : NEG;
      float t = fmaxf(v0, v1);
#pragma unroll
      for (int off = 8; off; off >>= 1) t = fmaxf(t, __shfl_xor(t, off, 32));
      float mn = fmaxf(m_i[i], t);
      float al = __expf(m_i[i] - mn);
      float p0 = __expf(v0 - mn);
      float p1 = __expf(v1 - mn);
      float rs = p0 + p1;
#pragma unroll
      for (int off = 8; off; off >>= 1) rs += __shfl_xor(rs, off, 32);
      l_i[i] = l_i[i] * al + rs;
      m_i[i] = mn;
      alpha8[i] = al;
      // C-layout -> LDS (row-major 16x32) for the A-fragment transpose
      p_lds[(i + 8 * h) * 32 + r]      = f2bf(p0);
      p_lds[(i + 8 * h) * 32 + 16 + r] = f2bf(p1);
    }

    // ---- P fragment (16x32 A-layout) from LDS ----------------------------
    FragB pf;
    {
      const unsigned short* pp = p_lds + r * 32 + h * 8;
      pf.q[0] = *(const uint4*)pp;
      pf.q[1] = *(const uint4*)(pp + 16);
    }

    // ---- O = alpha*O + P V  (16 n-tiles over hd, V from LDS) -------------
#pragma unroll
    for (int nt = 0; nt < 16; ++nt) {
      v8f t = o[nt];
#pragma unroll
      for (int i = 0; i < 8; ++i) t[i] *= alpha8[i];
      FragB vbf;
      const unsigned short* pv = vt_p + (size_t)(nt * 16 + r) * 32 + h * 8;
      vbf.q[0] = *(const uint4*)pv;
      vbf.q[1] = *(const uint4*)(pv + 16);
      o[nt] = wmma_bf16(pf, vbf, t);
    }
  }

  // ---- normalize and write ctx [B,S,D] bf16 ------------------------------
#pragma unroll
  for (int nt = 0; nt < 16; ++nt)
#pragma unroll
    for (int i = 0; i < 8; ++i) {
      int row = q0 + i + 8 * h;
      int col = hh * HDIM + nt * 16 + r;
      ctx[((size_t)(b * S + row)) * DMODEL + col] = f2bf(o[nt][i] / l_i[i]);
    }
}

// ---------------------------------------------------------------------------
extern "C" void kernel_launch(void* const* d_in, const int* in_sizes, int n_in,
                              void* d_out, int out_size, void* d_ws, size_t ws_size,
                              hipStream_t stream) {
  (void)in_sizes; (void)n_in; (void)out_size; (void)ws_size;
  const float* X  = (const float*)d_in[0];
  const float* wq = (const float*)d_in[1];
  const float* wk = (const float*)d_in[2];
  const float* wv = (const float*)d_in[3];
  const float* wo = (const float*)d_in[4];
  float* out = (float*)d_out;

  const int M = BATCH * SEQ;      // 4096
  const int N = DMODEL;           // 4096
  const int K = DMODEL;           // 4096
  const long n1 = (long)M * DMODEL;   // 16,777,216 elements per slab

  unsigned short* w  = (unsigned short*)d_ws;
  unsigned short* Xb  = w + 0L  * n1;
  unsigned short* Wqb = w + 1L  * n1;
  unsigned short* Wkb = w + 2L  * n1;
  unsigned short* Wvb = w + 3L  * n1;
  unsigned short* Wob = w + 4L  * n1;
  unsigned short* qb  = w + 5L  * n1;
  unsigned short* kb  = w + 6L  * n1;
  unsigned short* vb  = w + 7L  * n1;
  unsigned short* qt  = w + 8L  * n1;
  unsigned short* kt  = w + 9L  * n1;
  unsigned short* vt  = w + 10L * n1;
  unsigned short* ctx = qb;       // reuse: qb is dead after rope_transpose

  // 1) casts to bf16
  cast_bf16_kernel<<<4096, 256, 0, stream>>>(X,  Xb,  n1);
  cast_bf16_kernel<<<4096, 256, 0, stream>>>(wq, Wqb, n1);
  cast_bf16_kernel<<<4096, 256, 0, stream>>>(wk, Wkb, n1);
  cast_bf16_kernel<<<4096, 256, 0, stream>>>(wv, Wvb, n1);
  cast_bf16_kernel<<<4096, 256, 0, stream>>>(wo, Wob, n1);

  // 2) Q/K/V projections (NT GEMM, bf16 out)
  dim3 gg(N / 128, M / 128);
  gemm_nt_bf16<<<gg, 256, 0, stream>>>(Xb, Wqb, nullptr, qb, M, N, K);
  gemm_nt_bf16<<<gg, 256, 0, stream>>>(Xb, Wkb, nullptr, kb, M, N, K);
  gemm_nt_bf16<<<gg, 256, 0, stream>>>(Xb, Wvb, nullptr, vb, M, N, K);

  // 3) RoPE + transposes
  long nblk = (n1 + 255) / 256;
  rope_transpose_kernel<<<nblk, 256, 0, stream>>>(qb, qt, n1);
  rope_transpose_kernel<<<nblk, 256, 0, stream>>>(kb, kt, n1);
  vtrans_kernel<<<nblk, 256, 0, stream>>>(vb, vt, n1);

  // 4) flash attention: one wave per 16 query rows, TDM-fed K/V tiles
  dim3 ag(SEQ / 16, NHEADS, BATCH);
  flash_attn_kernel<<<ag, 32, 0, stream>>>(qt, kt, vt, ctx, SEQ);

  // 5) output projection -> fp32 d_out
  gemm_nt_bf16<<<gg, 256, 0, stream>>>(ctx, Wob, out, nullptr, M, N, K);
}
